// CNP_Net_81887846466121
// MI455X (gfx1250) — matrix operand order repaired and checked by reading, not compiled, not run on hardware
//
#include <hip/hip_runtime.h>
#include <hip/hip_bf16.h>
#include <math.h>

typedef __attribute__((ext_vector_type(16))) _Float16 v16h;
typedef __attribute__((ext_vector_type(8)))  float    v8f;

#define HALF_LOG_2PI 0.9189385332046727f

// d_ws layout (bytes):
//   0    : float  sum32[32]      encoder partial sums (relu layer-1 outputs)
//   128  : float  logsum         log-prob accumulator
//   256  : float  c0[64]         r @ g_W0[:128] + g_b0
//   512  : float  w128[64]       g_W0[128,:]
//   768  : f16    W1F[2][2][32][16]  g_W1 in WMMA B-fragment order (kc,nc,lane,half)
//   4864 : f16    W2F[32][16]        g_W2 in WMMA B-fragment order (lane,half)

__global__ void k_init(float* ws) {
    int t = threadIdx.x;
    if (t < 33) ws[t] = 0.0f;           // sum32[0..31] + logsum(idx 32)
}

// ---------------- Encoder: per-row 2->8->32 MLP, accumulate sum of relu outputs ----
__global__ void __launch_bounds__(256)
k_encoder(const float* __restrict__ O,
          const float* __restrict__ W0, const float* __restrict__ B0,
          const float* __restrict__ W1, const float* __restrict__ B1,
          float* __restrict__ ws, int n) {
    __shared__ float sW0[16], sB0[8], sW1[256], sB1[32];
    const int t = threadIdx.x;
    if (t < 16)  sW0[t] = W0[t];
    if (t < 8)   sB0[t] = B0[t];
    sW1[t] = W1[t];                      // 256 threads, 256 weights
    if (t < 32)  sB1[t] = B1[t];
    __syncthreads();

    float acc[32];
#pragma unroll
    for (int k = 0; k < 32; ++k) acc[k] = 0.0f;

    const int stride = gridDim.x * blockDim.x;
    for (int i = blockIdx.x * 256 + t; i < n; i += stride) {
        float o0 = O[2 * i], o1 = O[2 * i + 1];
        float h0[8];
#pragma unroll
        for (int j = 0; j < 8; ++j)
            h0[j] = fmaxf(fmaf(o1, sW0[8 + j], fmaf(o0, sW0[j], sB0[j])), 0.0f);
#pragma unroll
        for (int k = 0; k < 32; ++k) {
            float s = sB1[k];
#pragma unroll
            for (int j = 0; j < 8; ++j) s = fmaf(h0[j], sW1[j * 32 + k], s);
            acc[k] += fmaxf(s, 0.0f);
        }
    }
    // wave32 shuffle reduction, then one global atomic per component per wave
#pragma unroll
    for (int k = 0; k < 32; ++k) {
        float v = acc[k];
        for (int off = 16; off > 0; off >>= 1) v += __shfl_down(v, off, 32);
        if ((t & 31) == 0) atomicAdd(&ws[k], v);
    }
}

// ---------------- Prep: r, c0, w128 and f16 WMMA B-fragment packing ----------------
__global__ void k_prep(const float* __restrict__ hW2, const float* __restrict__ hB2,
                       const float* __restrict__ gW0, const float* __restrict__ gB0,
                       const float* __restrict__ gW1, const float* __restrict__ gW2,
                       float* __restrict__ ws, int nctx) {
    __shared__ float meanh[32];
    __shared__ float r[128];
    const int t = threadIdx.x;          // 128 threads
    if (t < 32) meanh[t] = ws[t] * (1.0f / (float)nctx);
    __syncthreads();
    {   // r = meanh @ h_W2 + h_b2   (mean commutes with the linear last layer)
        float s = hB2[t];
        for (int k = 0; k < 32; ++k) s = fmaf(meanh[k], hW2[k * 128 + t], s);
        r[t] = s;
    }
    __syncthreads();
    if (t < 64) {                       // c0 = r @ g_W0[:128] + g_b0 ; w128 = g_W0[128,:]
        float s = gB0[t];
        for (int k = 0; k < 128; ++k) s = fmaf(r[k], gW0[k * 64 + t], s);
        ws[64 + t]  = s;
        ws[128 + t] = gW0[128 * 64 + t];
    }
    // pack g_W1 (64x32) into WMMA 16-bit B fragments: N=lane%16, K=(h&7)+2*(h&8)+8*(lane/16)
    // (same K order as the A fragments), two K-chunks, two N-chunks
    _Float16* W1F = (_Float16*)((char*)ws + 768);
    for (int idx = t; idx < 2048; idx += 128) {
        int h = idx & 15, lane = (idx >> 4) & 31, nc = (idx >> 9) & 1, kc = idx >> 10;
        int k = kc * 32 + (h & 7) + ((h & 8) << 1) + ((lane & 16) >> 1);
        int n = nc * 16 + (lane & 15);
        W1F[idx] = (_Float16)gW1[k * 32 + n];
    }
    // pack g_W2 (32x16) into one B fragment
    _Float16* W2F = (_Float16*)((char*)ws + 4864);
    for (int idx = t; idx < 512; idx += 128) {
        int h = idx & 15, lane = idx >> 4;
        int k = (h & 7) + ((h & 8) << 1) + ((lane & 16) >> 1);
        int n = lane & 15;
        W2F[idx] = (_Float16)gW2[k * 16 + n];
    }
}

// ---------------- Decoder: WMMA for 64->32 and 32->16, scalar tail ------------------
__global__ void __launch_bounds__(256)
k_decoder(const float* __restrict__ T,
          const float* __restrict__ gB1, const float* __restrict__ gB2,
          const float* __restrict__ gW3, const float* __restrict__ gB3,
          const float* __restrict__ gW4, const float* __restrict__ gB4,
          const float* __restrict__ ws, float* __restrict__ out,
          float* __restrict__ logsum, int ntgt) {
    // per-wave slices; all producer/consumer pairs are intra-wave (DS in-order per wave),
    // so no block barriers are required anywhere in this kernel.
    __shared__ _Float16 a0h[8][32][64];   // 32 KB: layer-0 activations (f16, A-side rows)
    __shared__ _Float16 a1h[8][32][32];   // 16 KB: layer-1 activations
    const int t = threadIdx.x;
    const int w = t >> 5, lane = t & 31;
    const int rowL = w * 32 + lane;       // local row within block (one row per thread)
    const int row  = blockIdx.x * 256 + rowL;
    const int rowC = row < ntgt ? row : ntgt - 1;   // clamp for full-EXEC WMMA tiles

    const float* c0   = ws + 64;
    const float* w128 = ws + 128;
    const float  tx   = T[2 * rowC];

    // phase 1: layer0 per lane: a0 = relu(c0 + tx * g_W0[128,:]), store f16 rows to LDS
#pragma unroll 8
    for (int j = 0; j < 64; ++j)
        a0h[w][lane][j] = (_Float16)fmaxf(fmaf(tx, w128[j], c0[j]), 0.0f);

    const _Float16* W1F = (const _Float16*)((const char*)ws + 768);
    const _Float16* W2F = (const _Float16*)((const char*)ws + 4864);
    const int koff = (lane & 16) ? 8 : 0;      // A-fragment K offset for upper half-wave
    const int m    = lane & 15;
    const int hi8  = (lane & 16) >> 1;          // +8 rows for lanes 16..31 in C/D layout

    // phase 2: layer1 (64->32) as 2 K-chunks x 2 N-chunks of v_wmma_f32_16x16x32_f16
#pragma unroll
    for (int tile = 0; tile < 2; ++tile) {
        const int rloc = tile * 16 + m;
        union { v16h v; unsigned u[8]; } aA[2];
#pragma unroll
        for (int kc = 0; kc < 2; ++kc)
#pragma unroll
            for (int vv = 0; vv < 8; ++vv) {
                // 16-bit A 16x32 layout: VGPR v -> K = 2*(v%4) + 16*(v/4) (+8 upper lanes)
                int kl = kc * 32 + ((vv & 3) << 1) + ((vv >> 2) << 4) + koff;
                aA[kc].u[vv] = *(const unsigned*)&a0h[w][rloc][kl];
            }
#pragma unroll
        for (int nc = 0; nc < 2; ++nc) {
            v16h bf0 = *(const v16h*)(W1F + ((0 * 2 + nc) * 32 + lane) * 16);
            v16h bf1 = *(const v16h*)(W1F + ((1 * 2 + nc) * 32 + lane) * 16);
            float bias = gB1[nc * 16 + m];
            v8f c;
#pragma unroll
            for (int d = 0; d < 8; ++d) c[d] = bias;
            c = __builtin_amdgcn_wmma_f32_16x16x32_f16(false, aA[0].v, false, bf0,
                                                       (short)0, c, false, false);
            c = __builtin_amdgcn_wmma_f32_16x16x32_f16(false, aA[1].v, false, bf1,
                                                       (short)0, c, false, false);
#pragma unroll
            for (int d = 0; d < 8; ++d) {
                int rr = tile * 16 + d + (hi8 >> 0 ? (lane & 16 ? 8 : 0) : 0);
                rr = tile * 16 + d + ((lane & 16) ? 8 : 0);
                a1h[w][rr][nc * 16 + m] = (_Float16)fmaxf(c[d], 0.0f);
            }
        }
    }

    // phase 3: layer2 (32->16), one WMMA per tile; reuse this wave's a0 slice as f32 scratch
    float* a2f = (float*)&a0h[w][0][0];   // 4 KB slice, need 32*16*4 = 2 KB (same wave only)
#pragma unroll
    for (int tile = 0; tile < 2; ++tile) {
        const int rloc = tile * 16 + m;
        union { v16h v; unsigned u[8]; } aA;
#pragma unroll
        for (int vv = 0; vv < 8; ++vv) {
            int kl = ((vv & 3) << 1) + ((vv >> 2) << 4) + koff;
            aA.u[vv] = *(const unsigned*)&a1h[w][rloc][kl];
        }
        v16h bf = *(const v16h*)(W2F + lane * 16);
        float bias = gB2[m];
        v8f c;
#pragma unroll
        for (int d = 0; d < 8; ++d) c[d] = bias;
        c = __builtin_amdgcn_wmma_f32_16x16x32_f16(false, aA.v, false, bf,
                                                   (short)0, c, false, false);
#pragma unroll
        for (int d = 0; d < 8; ++d) {
            int rr = tile * 16 + d + ((lane & 16) ? 8 : 0);
            a2f[rr * 16 + m] = fmaxf(c[d], 0.0f);
        }
    }

    // phase 4: scalar tail 16->8->2, phi store, log-prob reduction (divergence OK now)
    float lp = 0.0f;
    if (row < ntgt) {
        float a2[16];
#pragma unroll
        for (int i = 0; i < 16; ++i) a2[i] = a2f[lane * 16 + i];
        float h3[8];
#pragma unroll
        for (int j = 0; j < 8; ++j) {
            float s = gB3[j];
#pragma unroll
            for (int i = 0; i < 16; ++i) s = fmaf(a2[i], gW3[i * 8 + j], s);
            h3[j] = fmaxf(s, 0.0f);
        }
        float p0 = gB4[0], p1 = gB4[1];
#pragma unroll
        for (int j = 0; j < 8; ++j) {
            p0 = fmaf(h3[j], gW4[2 * j], p0);
            p1 = fmaf(h3[j], gW4[2 * j + 1], p1);
        }
        out[2 * row]     = p0;
        out[2 * row + 1] = p1;
        float sig = (p1 > 20.0f) ? p1 : log1pf(__expf(p1));
        float y   = T[2 * row + 1];
        float z   = (y - p0) / sig;
        lp = fmaf(-0.5f * z, z, -__logf(sig)) - HALF_LOG_2PI;
    }
    for (int off = 16; off > 0; off >>= 1) lp += __shfl_down(lp, off, 32);
    if (lane == 0) atomicAdd(logsum, lp);
}

__global__ void k_final(const float* logsum, float* dst, float inv) {
    *dst = *logsum * inv;
}

extern "C" void kernel_launch(void* const* d_in, const int* in_sizes, int n_in,
                              void* d_out, int out_size, void* d_ws, size_t ws_size,
                              hipStream_t stream) {
    const float* O   = (const float*)d_in[0];
    const float* T   = (const float*)d_in[1];
    const float* hW0 = (const float*)d_in[2];
    const float* hB0 = (const float*)d_in[3];
    const float* hW1 = (const float*)d_in[4];
    const float* hB1 = (const float*)d_in[5];
    const float* hW2 = (const float*)d_in[6];
    const float* hB2 = (const float*)d_in[7];
    const float* gW0 = (const float*)d_in[8];
    const float* gB0 = (const float*)d_in[9];
    const float* gW1 = (const float*)d_in[10];
    const float* gB1 = (const float*)d_in[11];
    const float* gW2 = (const float*)d_in[12];
    const float* gB2 = (const float*)d_in[13];
    const float* gW3 = (const float*)d_in[14];
    const float* gB3 = (const float*)d_in[15];
    const float* gW4 = (const float*)d_in[16];
    const float* gB4 = (const float*)d_in[17];

    const int nctx = in_sizes[0] / 2;
    const int ntgt = in_sizes[1] / 2;
    float* ws  = (float*)d_ws;
    float* out = (float*)d_out;

    k_init<<<1, 64, 0, stream>>>(ws);
    k_encoder<<<200, 256, 0, stream>>>(O, hW0, hB0, hW1, hB1, ws, nctx);
    k_prep<<<1, 128, 0, stream>>>(hW2, hB2, gW0, gB0, gW1, gW2, ws, nctx);
    k_decoder<<<(ntgt + 255) / 256, 256, 0, stream>>>(T, gB1, gB2, gW3, gB3, gW4, gB4,
                                                      ws, out, ws + 32, ntgt);
    k_final<<<1, 1, 0, stream>>>(ws + 32, out + 2 * (size_t)ntgt, 1.0f / (float)ntgt);
}
